// GNNRegressor_5059471475392
// MI455X (gfx1250) — matrix operand order repaired and checked by reading, not compiled, not run on hardware
//
#include <hip/hip_runtime.h>

typedef __attribute__((ext_vector_type(16))) _Float16 v16h;
typedef __attribute__((ext_vector_type(4)))  _Float16 v4h;
typedef __attribute__((ext_vector_type(8)))  float    v8f;
typedef __attribute__((ext_vector_type(4)))  float    v4f;

#define TILE_M 64
#define TILE_N 64
#define TILE_K 32

// C[M,N] = act(A[M,K] @ B[K,N] + bias), row-major f32 in/out, f16 WMMA inside,
// optional batching via blockIdx.z strides.
// CONTRACT: M % 64 == 0 and N % 64 == 0 (all call sites satisfy this).
// EDGEK=false additionally requires K % 32 == 0 (and hence 16B-aligned A rows).
// LDS tiles are staged directly in WMMA *fragment order*: per 16-wide tile,
// per lane, 16 contiguous halves -> fragment load = one v16h LDS read
// (2x ds_load_b128).
template <bool EDGEK>
__global__ __launch_bounds__(128)
void wmma_gemm_kernel(const float* __restrict__ A, const float* __restrict__ Bm,
                      const float* __restrict__ bias, float* __restrict__ C,
                      int M, int N, int K,
                      long long strideA, long long strideB, long long strideC,
                      int relu)
{
    // [tile(4)][lane(32)] -> one v16h fragment slice per lane
    __shared__ v16h lAf[4 * 32];   // 4 row-tiles of A
    __shared__ v16h lBf[4 * 32];   // 4 col-tiles of B

    const int tid   = threadIdx.x;
    const int wave  = tid >> 5;
    const int lane  = tid & 31;
    const int lhalf = lane & 15;  // 0..15
    const int lgrp  = lane >> 4;  // 0 or 1

    const int rowBase = blockIdx.x * TILE_M;
    const int colBase = blockIdx.y * TILE_N;
    const long long zb = blockIdx.z;
    A  += zb * strideA;
    Bm += zb * strideB;
    C  += zb * strideC;

    // Init accumulators with bias (each lane's 8 C values share one column)
    v8f acc[4];
#pragma unroll
    for (int nb = 0; nb < 4; ++nb) {
        float bv = bias ? bias[colBase + (nb << 4) + lhalf] : 0.f;
        v8f a;
#pragma unroll
        for (int e = 0; e < 8; ++e) a[e] = bv;
        acc[nb] = a;
    }

    _Float16* pA = (_Float16*)lAf;
    _Float16* pB = (_Float16*)lBf;

    for (int k0 = 0; k0 < K; k0 += TILE_K) {
        // ---- Stage A tile (64x32) as fragments; quads along K, packed 8B stores
        for (int q = tid; q < (TILE_M * TILE_K / 4); q += 128) {
            int m  = q >> 3;              // 8 k-quads per row
            int qk = (q & 7) << 2;        // 0,4,...,28
            int gr = rowBase + m;
            int g   = qk >> 4;                              // A layout inverse
            int hi0 = qk & 15;
            int fl  = (m & 15) + (g << 4);
            int rowTile = m >> 4;
            v4h pk;
            if (EDGEK) {
#pragma unroll
                for (int t = 0; t < 4; ++t) {
                    int gk = k0 + qk + t;
                    float x = (gk < K) ? A[(long long)gr * K + gk] : 0.f;
                    pk[t] = (_Float16)x;
                }
            } else {
                v4f xv = *(const v4f*)(A + (long long)gr * K + k0 + qk);
#pragma unroll
                for (int t = 0; t < 4; ++t) pk[t] = (_Float16)xv[t];
            }
            *(v4h*)(pA + (((rowTile << 5) + fl) << 4) + hi0) = pk;
        }
        // ---- Stage B tile (32x64) as fragments; quads along K, coalesced in N
        for (int q = tid; q < (TILE_K * TILE_N / 4); q += 128) {
            int n  = q & 63;              // consecutive tid -> consecutive n
            int kq = (q >> 6) << 2;       // 0,4,...,28
            int gn = colBase + n;
            int g   = kq >> 4;                              // B layout inverse
            int hi0 = kq & 15;
            int fl  = (n & 15) + (g << 4);
            int colTile = n >> 4;
            v4h pk;
#pragma unroll
            for (int t = 0; t < 4; ++t) {
                int gk = k0 + kq + t;
                float x = (!EDGEK || gk < K) ? Bm[(long long)gk * N + gn] : 0.f;
                pk[t] = (_Float16)x;
            }
            *(v4h*)(pB + (((colTile << 5) + fl) << 4) + hi0) = pk;
        }
        __syncthreads();

        // ---- Fragments are contiguous: one v16h LDS read each
        v16h afrag = lAf[(wave << 5) + lane];
#pragma unroll
        for (int nb = 0; nb < 4; ++nb) {
            v16h bfrag = lBf[(nb << 5) + lane];
            acc[nb] = __builtin_amdgcn_wmma_f32_16x16x32_f16(
                false, afrag, false, bfrag, (short)0, acc[nb], false, false);
        }
        __syncthreads();
    }

    // Store: C/D layout m = v + 8*lgrp, n = lane&15 (no edge: M%64==N%64==0)
#pragma unroll
    for (int nb = 0; nb < 4; ++nb) {
        int gn = colBase + (nb << 4) + lhalf;
#pragma unroll
        for (int vv = 0; vv < 8; ++vv) {
            int gm = rowBase + (wave << 4) + vv + (lgrp << 3);
            float r = acc[nb][vv];
            if (relu) r = r > 0.f ? r : 0.f;
            C[(long long)gm * N + gn] = r;
        }
    }
}

// One wave per row: s1[r] = h[r,:]·a1, s2[r] = h[r,:]·a2 (coalesced reads)
__global__ __launch_bounds__(128)
void s1s2_kernel(const float* __restrict__ h, const float* __restrict__ a1,
                 const float* __restrict__ a2, float* __restrict__ s1,
                 float* __restrict__ s2, int rows, int Hd)
{
    int r    = blockIdx.x * 4 + (threadIdx.x >> 5);
    int lane = threadIdx.x & 31;
    if (r >= rows) return;
    const float* hr = h + (long long)r * Hd;
    float d1 = 0.f, d2 = 0.f;
    for (int k = lane; k < Hd; k += 32) {
        float x = hr[k];
        d1 += x * a1[k];
        d2 += x * a2[k];
    }
#pragma unroll
    for (int o = 16; o > 0; o >>= 1) {
        d1 += __shfl_xor(d1, o, 32);
        d2 += __shfl_xor(d2, o, 32);
    }
    if (lane == 0) { s1[r] = d1; s2[r] = d2; }
}

// One block per (b,i) row: leaky_relu(s1_i + s2_j), mask, softmax over j
__global__ __launch_bounds__(128)
void attn_kernel(const float* __restrict__ s1, const float* __restrict__ s2,
                 const int* __restrict__ adj, float* __restrict__ att, int Nn)
{
    __shared__ float red[128];
    int j  = threadIdx.x;
    int bi = blockIdx.x;        // b*N + i
    int b  = bi / Nn;

    float e = s1[bi] + s2[b * Nn + j];
    e = e > 0.f ? e : 0.2f * e;
    float v = (adj[(long long)bi * Nn + j] > 0) ? e : -9.0e15f;

    red[j] = v; __syncthreads();
    for (int s = 64; s > 0; s >>= 1) { if (j < s) red[j] = fmaxf(red[j], red[j + s]); __syncthreads(); }
    float mx = red[0]; __syncthreads();

    float ex = __expf(v - mx);
    red[j] = ex; __syncthreads();
    for (int s = 64; s > 0; s >>= 1) { if (j < s) red[j] += red[j + s]; __syncthreads(); }
    float sum = red[0];

    att[(long long)bi * Nn + j] = ex / sum;
}

__global__ void meanpool_kernel(const float* __restrict__ X, float* __restrict__ graph,
                                int Nn, int Hd)
{
    int b = blockIdx.x, hh = threadIdx.x;
    const float* xb = X + (long long)b * Nn * Hd;
    float s = 0.f;
    for (int n = 0; n < Nn; ++n) s += xb[n * Hd + hh];
    graph[b * Hd + hh] = s * (1.0f / Nn);
}

__global__ void small_linear_kernel(const float* __restrict__ A, const float* __restrict__ W,
                                    const float* __restrict__ bias, float* __restrict__ C,
                                    int M, int K, int N, int relu)
{
    int idx = blockIdx.x * blockDim.x + threadIdx.x;
    if (idx >= M * N) return;
    int r = idx / N, c = idx % N;
    float s = bias ? bias[c] : 0.f;
    const float* ar = A + r * K;
    for (int k = 0; k < K; ++k) s += ar[k] * W[k * N + c];
    if (relu) s = s > 0.f ? s : 0.f;
    C[idx] = s;
}

__global__ void concat_kernel(const float* __restrict__ a, const float* __restrict__ b,
                              float* __restrict__ c, int M, int Na, int Nb)
{
    int idx = blockIdx.x * blockDim.x + threadIdx.x;
    int Nt = Na + Nb;
    if (idx >= M * Nt) return;
    int r = idx / Nt, cc = idx % Nt;
    c[idx] = (cc < Na) ? a[r * Na + cc] : b[r * Nb + (cc - Na)];
}

extern "C" void kernel_launch(void* const* d_in, const int* in_sizes, int n_in,
                              void* d_out, int out_size, void* d_ws, size_t ws_size,
                              hipStream_t stream)
{
    (void)in_sizes; (void)n_in; (void)out_size; (void)ws_size;
    const int Bb = 32, Nn = 128, Ff = 78, Hh = 256, Ss = 20, Tt = 13, Ll = 3;
    const int BN = Bb * Nn;                 // 4096

    const float* nf   = (const float*)d_in[0];
    // d_in[1] = edge_features: unused by the reference
    const int*   adj  = (const int*)  d_in[2];
    const float* scaf = (const float*)d_in[3];
    const float* Wn   = (const float*)d_in[4];
    const float* bn   = (const float*)d_in[5];
    const float* gatW = (const float*)d_in[6];
    const float* gatA = (const float*)d_in[7];
    const float* Wsc  = (const float*)d_in[8];
    const float* bsc  = (const float*)d_in[9];
    const float* gpw1 = (const float*)d_in[10];
    const float* gpb1 = (const float*)d_in[11];
    const float* gpw2 = (const float*)d_in[12];
    const float* gpb2 = (const float*)d_in[13];
    const float* ow1  = (const float*)d_in[14];
    const float* ob1  = (const float*)d_in[15];
    const float* ow2  = (const float*)d_in[16];
    const float* ob2  = (const float*)d_in[17];
    float* out = (float*)d_out;

    // Workspace carve-up (floats); total ~2.66M floats ~ 10.6 MB
    float* ws    = (float*)d_ws;
    float* X     = ws;                         // BN*H = 1048576
    float* Hb    = X    + (size_t)BN * Hh;     // 1048576
    float* S1    = Hb   + (size_t)BN * Hh;     // 4096
    float* S2    = S1   + BN;                  // 4096
    float* ATT   = S2   + BN;                  // B*N*N = 524288
    float* graph = ATT  + (size_t)Bb * Nn * Nn;// 8192
    float* t1    = graph+ Bb * Hh;             // 4096
    float* g     = t1   + Bb * (Hh / 2);       // 2048
    float* z0    = g    + Bb * (Hh / 4);       // 8192
    float* t2    = z0   + Bb * Hh;             // 4096
    float* sed   = t2   + Bb * (Hh / 2);       // 2048
    float* cbuf  = sed  + Bb * (Hh / 4);       // 4096
    float* hdn   = cbuf + Bb * (Hh / 2);       // 4096

    // 1) Node embed: X = nf(4096x78) @ W_node(78x256) + b_node  [WMMA, ragged K]
    wmma_gemm_kernel<true><<<dim3(BN / TILE_M, Hh / TILE_N, 1), 128, 0, stream>>>(
        nf, Wn, bn, X, BN, Hh, Ff, 0, 0, 0, 0);

    // 2) GAT layers
    for (int l = 0; l < Ll; ++l) {
        // h = X @ gat_W[l]  (4096x256 @ 256x256)   [WMMA, clean dims]
        wmma_gemm_kernel<false><<<dim3(BN / TILE_M, Hh / TILE_N, 1), 128, 0, stream>>>(
            X, gatW + (size_t)l * Hh * Hh, nullptr, Hb, BN, Hh, Hh, 0, 0, 0, 0);
        // s1 = h@a1, s2 = h@a2 (wave-per-row)
        s1s2_kernel<<<BN / 4, 128, 0, stream>>>(
            Hb, gatA + (size_t)l * 2 * Hh, gatA + (size_t)l * 2 * Hh + Hh, S1, S2, BN, Hh);
        // masked softmax attention rows
        attn_kernel<<<BN, Nn, 0, stream>>>(S1, S2, adj, ATT, Nn);
        // X = relu(att @ h) batched per b: 128x128 @ 128x256   [WMMA, clean dims]
        wmma_gemm_kernel<false><<<dim3(Nn / TILE_M, Hh / TILE_N, Bb), 128, 0, stream>>>(
            ATT, Hb, nullptr, X, Nn, Hh, Nn,
            (long long)Nn * Nn, (long long)Nn * Hh, (long long)Nn * Hh, 1);
    }

    // 3) graph = mean over nodes
    meanpool_kernel<<<Bb, Hh, 0, stream>>>(X, graph, Nn, Hh);

    // 4) g = pool(graph)
    small_linear_kernel<<<(Bb * (Hh / 2) + 255) / 256, 256, 0, stream>>>(
        graph, gpw1, gpb1, t1, Bb, Hh, Hh / 2, 1);
    small_linear_kernel<<<(Bb * (Hh / 4) + 255) / 256, 256, 0, stream>>>(
        t1, gpw2, gpb2, g, Bb, Hh / 2, Hh / 4, 1);

    // 5) s = pool(scaffold @ W_sc + b_sc)
    small_linear_kernel<<<(Bb * Hh + 255) / 256, 256, 0, stream>>>(
        scaf, Wsc, bsc, z0, Bb, Ss, Hh, 0);
    small_linear_kernel<<<(Bb * (Hh / 2) + 255) / 256, 256, 0, stream>>>(
        z0, gpw1, gpb1, t2, Bb, Hh, Hh / 2, 1);
    small_linear_kernel<<<(Bb * (Hh / 4) + 255) / 256, 256, 0, stream>>>(
        t2, gpw2, gpb2, sed, Bb, Hh / 2, Hh / 4, 1);

    // 6) c = concat(g, s); hdn = relu(c @ out_w1 + b1); out = hdn @ out_w2 + b2
    concat_kernel<<<(Bb * (Hh / 2) + 255) / 256, 256, 0, stream>>>(
        g, sed, cbuf, Bb, Hh / 4, Hh / 4);
    small_linear_kernel<<<(Bb * (Hh / 2) + 255) / 256, 256, 0, stream>>>(
        cbuf, ow1, ob1, hdn, Bb, Hh / 2, Hh / 2, 1);
    small_linear_kernel<<<(Bb * Tt + 255) / 256, 256, 0, stream>>>(
        hdn, ow2, ob2, out, Bb, Hh / 2, Tt, 0);
}
